// GraphSAGE_16415365005404
// MI455X (gfx1250) — matrix operand order repaired and checked by reading, not compiled, not run on hardware
//
#include <hip/hip_runtime.h>

// ---------------------------------------------------------------------------
// GraphSAGE (3 x SAGEConv + BN + ReLU) for gfx1250 (MI455X), wave32 + WMMA.
//   agg = mean over in-edges of h[src];  out = agg @ w_l + h @ w_r + b
// Activations are kept in bf16 between stages; the WMMA GEMM inner loop is
// pure uint4 loads + v_wmma_f32_16x16x32_bf16 (fp32 accumulate). Aggregation
// uses fp32 global atomics into an L2-resident accumulator.
// ---------------------------------------------------------------------------

typedef __bf16 bf16_t;
typedef bf16_t bf16x16 __attribute__((ext_vector_type(16)));
typedef float  f32x8   __attribute__((ext_vector_type(8)));

__device__ __forceinline__ unsigned short f2bf_rne(float f) {
    unsigned int u = __float_as_uint(f);
    unsigned int r = (u + 0x7FFFu + ((u >> 16) & 1u)) >> 16;
    return (unsigned short)r;
}

__device__ __forceinline__ float bf2f(unsigned short s) {
    return __uint_as_float((unsigned int)s << 16);
}

// Build one 16-element bf16 fragment from two 16B chunks (A-layout: the two
// chunks are K[8h..8h+7] and K[16+8h..16+8h+7]; B-layout: the same, p1 = p0 + 16).
__device__ __forceinline__ bf16x16 load_frag(const unsigned short* __restrict__ p0,
                                             const unsigned short* __restrict__ p1) {
    union { uint4 q[2]; bf16x16 v; } u;
    u.q[0] = *(const uint4*)p0;
    u.q[1] = *(const uint4*)p1;
    return u.v;
}

// ---------------------------------------------------------------------------
__global__ void sage_fill0(float* __restrict__ p, int n) {
    int i = blockIdx.x * blockDim.x + threadIdx.x;
    if (i < n) p[i] = 0.0f;
}

__global__ void sage_deg_count(const int* __restrict__ ei, float* __restrict__ deg, int E) {
    int i = blockIdx.x * blockDim.x + threadIdx.x;
    if (i < E) atomicAdd(&deg[ei[E + i]], 1.0f);
}

__global__ void sage_inv_deg(float* __restrict__ deg, int n) {
    int i = blockIdx.x * blockDim.x + threadIdx.x;
    if (i < n) deg[i] = 1.0f / fmaxf(deg[i], 1.0f);
}

// wT[n*K + k] = bf16(w[k*Nc + n])   (w is K x Nc row-major)
__global__ void sage_cvt_wT(const float* __restrict__ w, unsigned short* __restrict__ wT,
                            int K, int Nc) {
    int idx = blockIdx.x * blockDim.x + threadIdx.x;
    if (idx >= K * Nc) return;
    int n = idx / K;
    int k = idx - n * K;
    wT[idx] = f2bf_rne(w[(size_t)k * Nc + n]);
}

// Plain fp32 -> bf16 copy (for the input features x).
__global__ void sage_cvt_bf16(const float* __restrict__ src, unsigned short* __restrict__ dst,
                              int n) {
    int i = blockIdx.x * blockDim.x + threadIdx.x;
    if (i < n) dst[i] = f2bf_rne(src[i]);
}

// agg fp32 * invdeg[row] -> bf16 (fuses the mean divide into the conversion).
__global__ void sage_cvt_agg(const float* __restrict__ agg, const float* __restrict__ invdeg,
                             unsigned short* __restrict__ aggb, int n, int D) {
    int i = blockIdx.x * blockDim.x + threadIdx.x;
    if (i >= n) return;
    int row = i / D;
    aggb[i] = f2bf_rne(agg[i] * invdeg[row]);
}

// Scatter-add: gathers bf16 rows (half the HBM traffic), accumulates fp32 in L2.
// Consecutive threads cover consecutive 4-elem feature chunks of one edge.
__global__ void sage_scatter(const unsigned short* __restrict__ h, const int* __restrict__ ei,
                             float* __restrict__ agg, int E, int D) {
    int chunks = D >> 2;
    int idx = blockIdx.x * blockDim.x + threadIdx.x;
    if (idx >= E * chunks) return;
    int e = idx / chunks;
    int c = (idx - e * chunks) << 2;
    int s = ei[e];
    int d = ei[E + e];
    ushort4 v = *(const ushort4*)(h + (size_t)s * D + c);
    float* o = agg + (size_t)d * D + c;
    atomicAdd(o + 0, bf2f(v.x));
    atomicAdd(o + 1, bf2f(v.y));
    atomicAdd(o + 2, bf2f(v.z));
    atomicAdd(o + 3, bf2f(v.w));
}

// Fused SAGE GEMM: out = aggb @ wlT' + hb @ wrT' + b, optional BN(eval)+ReLU.
// One 16x16 C tile per wave32; 8 waves per 256-thread block. Inner loop is
// 8x uint4 loads + 2x v_wmma_f32_16x16x32_bf16 per 32-K step.
__global__ __launch_bounds__(256) void sage_gemm_wmma(
    const unsigned short* __restrict__ Aagg, const unsigned short* __restrict__ Ah,
    const unsigned short* __restrict__ wlT, const unsigned short* __restrict__ wrT,
    const float* __restrict__ bias,
    const float* __restrict__ gamma, const float* __restrict__ beta,
    const float* __restrict__ mean, const float* __restrict__ var, int has_bn,
    unsigned short* __restrict__ out_bf, float* __restrict__ out_f32,
    int Hc, int K, int tilesN, int totalTiles) {
    int wave = threadIdx.x >> 5;
    int lane = threadIdx.x & 31;
    int tile = blockIdx.x * 8 + wave;
    if (tile >= totalTiles) return;  // wave-uniform: EXEC stays all-ones for WMMA

    int tm = (tile / tilesN) * 16;
    int tn = (tile - (tile / tilesN) * tilesN) * 16;
    int l15 = lane & 15;
    int half = lane >> 4;
    int m = tm + l15;  // A row this lane carries
    int n = tn + l15;  // B col / C col this lane carries

    // A-layout per lane: chunk0 = K[kt+8h .. +7], chunk1 = K[kt+16+8h .. +7].
    const unsigned short* ag = Aagg + (size_t)m * K + 8 * half;
    const unsigned short* ah = Ah + (size_t)m * K + 8 * half;
    // B-layout per lane: chunk0 = K[kt+8h .. +7], chunk1 = K[kt+16+8h .. +7].
    const unsigned short* bl = wlT + (size_t)n * K + 8 * half;
    const unsigned short* br = wrT + (size_t)n * K + 8 * half;

    f32x8 acc = {};
    for (int kt = 0; kt < K; kt += 32) {
        bf16x16 a_g = load_frag(ag + kt, ag + kt + 16);
        bf16x16 a_h = load_frag(ah + kt, ah + kt + 16);
        bf16x16 b_l = load_frag(bl + kt, bl + kt + 16);
        bf16x16 b_r = load_frag(br + kt, br + kt + 16);
        acc = __builtin_amdgcn_wmma_f32_16x16x32_bf16(false, a_g, false, b_l,
                                                      (short)0, acc, false, false);
        acc = __builtin_amdgcn_wmma_f32_16x16x32_bf16(false, a_h, false, b_r,
                                                      (short)0, acc, false, false);
    }

    // Epilogue. C/D layout: VGPR r -> row tm + 8*half + r, col = tn + (lane&15).
    float bb = bias[n];
    float sc = 1.0f, sh = 0.0f;
    if (has_bn) {
        float g = gamma[n], be = beta[n], mu = mean[n], vv = var[n];
        sc = g * rsqrtf(vv + 1e-5f);
        sh = be - mu * sc;
    }
#pragma unroll
    for (int r = 0; r < 8; ++r) {
        int row = tm + 8 * half + r;
        float y = acc[r] + bb;
        if (has_bn) y = fmaxf(y * sc + sh, 0.0f);
        if (out_bf) out_bf[(size_t)row * Hc + n] = f2bf_rne(y);
        if (out_f32) out_f32[(size_t)row * Hc + n] = y;
    }
}

// ---------------------------------------------------------------------------
extern "C" void kernel_launch(void* const* d_in, const int* in_sizes, int n_in,
                              void* d_out, int out_size, void* d_ws, size_t ws_size,
                              hipStream_t stream) {
    const int D = 512, H = 512;
    const int N = in_sizes[0] / D;       // 10000 (divisible by 16)
    const int E = in_sizes[1] / 2;       // 160000
    const int O = in_sizes[16] / H;      // 256

    const float* x = (const float*)d_in[0];
    const int* ei = (const int*)d_in[1];
    const float* w_l1 = (const float*)d_in[2];
    const float* b1 = (const float*)d_in[3];
    const float* w_r1 = (const float*)d_in[4];
    const float* g1 = (const float*)d_in[5];
    const float* be1 = (const float*)d_in[6];
    const float* mu1 = (const float*)d_in[7];
    const float* v1 = (const float*)d_in[8];
    const float* w_l2 = (const float*)d_in[9];
    const float* b2 = (const float*)d_in[10];
    const float* w_r2 = (const float*)d_in[11];
    const float* g2 = (const float*)d_in[12];
    const float* be2 = (const float*)d_in[13];
    const float* mu2 = (const float*)d_in[14];
    const float* v2 = (const float*)d_in[15];
    const float* w_l3 = (const float*)d_in[16];
    const float* b3 = (const float*)d_in[17];
    const float* w_r3 = (const float*)d_in[18];

    // Workspace carve-up (~64 MB), all offsets 256B-aligned.
    char* w = (char*)d_ws;
    size_t off = 0;
    auto carve = [&](size_t bytes) {
        void* p = w + off;
        off = (off + bytes + 255) & ~(size_t)255;
        return p;
    };
    float* deg = (float*)carve((size_t)N * 4);
    float* agg = (float*)carve((size_t)N * D * 4);             // fp32 accumulator
    unsigned short* aggb = (unsigned short*)carve((size_t)N * D * 2);
    unsigned short* xb   = (unsigned short*)carve((size_t)N * D * 2);
    unsigned short* h1b  = (unsigned short*)carve((size_t)N * H * 2);
    unsigned short* h2b  = (unsigned short*)carve((size_t)N * H * 2);
    unsigned short* wl1T = (unsigned short*)carve((size_t)D * H * 2);
    unsigned short* wr1T = (unsigned short*)carve((size_t)D * H * 2);
    unsigned short* wl2T = (unsigned short*)carve((size_t)H * H * 2);
    unsigned short* wr2T = (unsigned short*)carve((size_t)H * H * 2);
    unsigned short* wl3T = (unsigned short*)carve((size_t)H * O * 2);
    unsigned short* wr3T = (unsigned short*)carve((size_t)H * O * 2);
    (void)ws_size;

    const int T = 256;
    auto blk = [&](long long n) { return (unsigned)((n + T - 1) / T); };

    // Degrees -> invdeg (shared by all layers; deterministic every call).
    sage_fill0<<<blk(N), T, 0, stream>>>(deg, N);
    sage_deg_count<<<blk(E), T, 0, stream>>>(ei, deg, E);
    sage_inv_deg<<<blk(N), T, 0, stream>>>(deg, N);

    // Weights -> bf16 transposed; x -> bf16.
    sage_cvt_wT<<<blk((long long)D * H), T, 0, stream>>>(w_l1, wl1T, D, H);
    sage_cvt_wT<<<blk((long long)D * H), T, 0, stream>>>(w_r1, wr1T, D, H);
    sage_cvt_wT<<<blk((long long)H * H), T, 0, stream>>>(w_l2, wl2T, H, H);
    sage_cvt_wT<<<blk((long long)H * H), T, 0, stream>>>(w_r2, wr2T, H, H);
    sage_cvt_wT<<<blk((long long)H * O), T, 0, stream>>>(w_l3, wl3T, H, O);
    sage_cvt_wT<<<blk((long long)H * O), T, 0, stream>>>(w_r3, wr3T, H, O);
    sage_cvt_bf16<<<blk((long long)N * D), T, 0, stream>>>(x, xb, N * D);

    const int chunks = D >> 2;
    const int tilesM = N / 16;

    // ----- Layer 1: agg(xb) -> GEMM+BN+ReLU -> h1b
    sage_fill0<<<blk((long long)N * D), T, 0, stream>>>(agg, N * D);
    sage_scatter<<<blk((long long)E * chunks), T, 0, stream>>>(xb, ei, agg, E, D);
    sage_cvt_agg<<<blk((long long)N * D), T, 0, stream>>>(agg, deg, aggb, N * D, D);
    {
        int tilesN = H / 16, total = tilesM * tilesN;
        sage_gemm_wmma<<<(total + 7) / 8, 256, 0, stream>>>(
            aggb, xb, wl1T, wr1T, b1, g1, be1, mu1, v1, 1, h1b, nullptr, H, D, tilesN, total);
    }

    // ----- Layer 2: agg(h1b) -> GEMM+BN+ReLU -> h2b
    sage_fill0<<<blk((long long)N * H), T, 0, stream>>>(agg, N * H);
    sage_scatter<<<blk((long long)E * chunks), T, 0, stream>>>(h1b, ei, agg, E, H);
    sage_cvt_agg<<<blk((long long)N * H), T, 0, stream>>>(agg, deg, aggb, N * H, H);
    {
        int tilesN = H / 16, total = tilesM * tilesN;
        sage_gemm_wmma<<<(total + 7) / 8, 256, 0, stream>>>(
            aggb, h1b, wl2T, wr2T, b2, g2, be2, mu2, v2, 1, h2b, nullptr, H, H, tilesN, total);
    }

    // ----- Layer 3: agg(h2b) -> GEMM (+bias only) -> d_out (fp32)
    sage_fill0<<<blk((long long)N * H), T, 0, stream>>>(agg, N * H);
    sage_scatter<<<blk((long long)E * chunks), T, 0, stream>>>(h2b, ei, agg, E, H);
    sage_cvt_agg<<<blk((long long)N * H), T, 0, stream>>>(agg, deg, aggb, N * H, H);
    {
        int tilesN = O / 16, total = tilesM * tilesN;
        sage_gemm_wmma<<<(total + 7) / 8, 256, 0, stream>>>(
            aggb, h2b, wl3T, wr3T, b3, nullptr, nullptr, nullptr, nullptr, 0,
            nullptr, (float*)d_out, O, H, tilesN, total);
    }
}
